// Encoder_17695265259992
// MI455X (gfx1250) — matrix operand order repaired and checked by reading, not compiled, not run on hardware
//
#include <hip/hip_runtime.h>
#include <hip/hip_bf16.h>

#define N_SETS 4096
#define DIM    128
#define DW     8
#define APAD   136   // padded LDS row stride (bf16 elems): 272B -> 68 dwords, staggers banks

typedef __attribute__((ext_vector_type(16))) __bf16 v16bf;
typedef __attribute__((ext_vector_type(8)))  float  v8f;

union Frag { uint4 u[2]; v16bf v; };

// round-to-nearest-even f32 -> bf16 (bit manip; avoids __bf16 arithmetic)
__device__ __forceinline__ unsigned short f2bf(float f) {
  unsigned u = __float_as_uint(f);
  u += 0x7FFFu + ((u >> 16) & 1u);
  return (unsigned short)(u >> 16);
}

__device__ __forceinline__ int lower_bound(const int* __restrict__ arr, int n, int val) {
  int lo = 0, hi = n;
  while (lo < hi) {
    int mid = (lo + hi) >> 1;
    if (arr[mid] < val) lo = mid + 1; else hi = mid;
  }
  return lo;
}

__global__ __launch_bounds__(32)
void encoder_kernel(const float* __restrict__ x,
                    const float* __restrict__ Wk,
                    const float* __restrict__ Wv,
                    const int*   __restrict__ batch,
                    float*       __restrict__ out,
                    int total) {
  __shared__ __attribute__((aligned(16))) unsigned short Abuf[16 * APAD];
  __shared__ __attribute__((aligned(16))) unsigned short Wbuf[16 * APAD];
  __shared__ __attribute__((aligned(16))) float          Ybuf[16 * 17];

  const int b    = blockIdx.x;
  const int lane = threadIdx.x;
  const int row  = lane & 15;   // M (A) / N (B) index
  const int hi   = lane >> 4;   // lane-group select

  // ---- stage combined weights W = [Wk ; Wv] (16 x 128) into LDS as bf16 ----
  for (int idx = lane; idx < 16 * DIM; idx += 32) {
    int n = idx >> 7, k = idx & 127;
    float w = (n < DW) ? Wk[n * DIM + k] : Wv[(n - DW) * DIM + k];
    Wbuf[n * APAD + k] = f2bf(w);
  }
  __syncthreads();

  // ---- B fragments (W^T, 32x16 per chunk), persistent across tiles ----
  // B layout (bf16, 32x16): lanes 0-15 hold K {0..7,16..23} (N=lane), lanes 16-31 K {8..15,24..31}.
  // B[k][n] = W[n][k] -> per lane two 8-bf16 runs of W row `row`.
  Frag fb[4];
  #pragma unroll
  for (int kc = 0; kc < 4; ++kc) {
    int kb = kc * 32 + hi * 8;
    fb[kc].u[0] = *(const uint4*)&Wbuf[row * APAD + kb];
    fb[kc].u[1] = *(const uint4*)&Wbuf[row * APAD + kb + 16];
  }

  // ---- segment bounds via binary search on sorted batch ids ----
  const int start = lower_bound(batch, total, b);
  const int end   = lower_bound(batch, total, b + 1);
  const int nseg  = end - start;

  // ---- PE frequencies for this lane's 4 columns (cols 4*lane .. 4*lane+3) ----
  const float lgc = -0.14391156831212787f;            // -ln(10000)/64
  const float f0  = __expf(lgc * (float)(2 * lane));
  const float f1  = __expf(lgc * (float)(2 * lane + 1));

  // each lane owns 2 of the 64 outer-product outputs: p = a*8 + c
  const int p0 = 2 * lane, p1 = 2 * lane + 1;
  const int a0 = p0 >> 3, c0 = p0 & 7;
  const int a1 = p1 >> 3, c1 = p1 & 7;
  float acc0 = 0.f, acc1 = 0.f;

  for (int base = 0; base < nseg; base += 16) {
    // ---- build xpos tile (16 x 128) in LDS as bf16; pad rows -> exact zero ----
    #pragma unroll 4
    for (int m = 0; m < 16; ++m) {
      float4 v = make_float4(0.f, 0.f, 0.f, 0.f);
      const int li = base + m;                         // position within segment
      if (li < nseg) {
        // coalesced: 32 lanes cover one full 512B row of x
        const float4 xv = *(const float4*)&x[(size_t)(start + li) * DIM + 4 * lane];
        const float pos = (float)li;
        const float ang0 = pos * f0, ang1 = pos * f1;
        v.x = xv.x + __sinf(ang0);
        v.y = xv.y + __cosf(ang0);
        v.z = xv.z + __sinf(ang1);
        v.w = xv.w + __cosf(ang1);
      }
      const unsigned short h0 = f2bf(v.x), h1 = f2bf(v.y);
      const unsigned short h2 = f2bf(v.z), h3 = f2bf(v.w);
      *(uint2*)&Abuf[m * APAD + 4 * lane] =
          make_uint2(((unsigned)h1 << 16) | h0, ((unsigned)h3 << 16) | h2);
    }
    if (base + 16 < nseg)   // pull next tile toward L2/L0 (global_prefetch_b8)
      __builtin_prefetch(&x[(size_t)(start + base + 16) * DIM + 4 * lane], 0, 3);
    __syncthreads();

    // ---- Y(16x16) = xpos(16x128) @ W^T(128x16), 4 chained bf16 WMMAs ----
    v8f acc = {0.f, 0.f, 0.f, 0.f, 0.f, 0.f, 0.f, 0.f};
    #pragma unroll
    for (int kc = 0; kc < 4; ++kc) {
      // A layout (bf16, 16x32): lane<16 row=lane, K {0..7,16..23}; lane>=16 K {8..15,24..31}
      Frag fa;
      const int ka = kc * 32 + hi * 8;
      fa.u[0] = *(const uint4*)&Abuf[row * APAD + ka];
      fa.u[1] = *(const uint4*)&Abuf[row * APAD + ka + 16];
      acc = __builtin_amdgcn_wmma_f32_16x16x32_bf16(
                false, fa.v, false, fb[kc].v, (short)0, acc, false, false);
    }

    // ---- D layout -> LDS: VGPR r = row (r | r+8), col = lane&15 ----
    #pragma unroll
    for (int r = 0; r < 8; ++r)
      Ybuf[(r + hi * 8) * 17 + row] = acc[r];
    __syncthreads();

    // ---- per-element outer product yv ⊗ yk, segment-summed in registers ----
    // Y cols 0..7 = yk, 8..15 = yv; padded rows are all-zero -> contribute 0.
    #pragma unroll
    for (int m = 0; m < 16; ++m) {
      const float* yr = &Ybuf[m * 17];
      acc0 += yr[8 + a0] * yr[c0];
      acc1 += yr[8 + a1] * yr[c1];
    }
    __syncthreads();   // protect Ybuf/Abuf before next tile overwrites them
  }

  // ---- write this set's 65 outputs (each element written exactly once) ----
  float* o = out + (size_t)b * (DW * DW + 1);
  o[p0] = acc0;
  o[p1] = acc1;
  if (lane == 0) o[DW * DW] = (float)nseg;
}

extern "C" void kernel_launch(void* const* d_in, const int* in_sizes, int n_in,
                              void* d_out, int out_size, void* d_ws, size_t ws_size,
                              hipStream_t stream) {
  const float* x     = (const float*)d_in[0];
  const float* Wk    = (const float*)d_in[1];
  const float* Wv    = (const float*)d_in[2];
  const int*   batch = (const int*)d_in[3];
  float*       out   = (float*)d_out;
  const int    total = in_sizes[3];   // 524288 ragged elements

  encoder_kernel<<<dim3(N_SETS), dim3(32), 0, stream>>>(x, Wk, Wv, batch, out, total);
}
